// CoA_86182813761681
// MI455X (gfx1250) — matrix operand order repaired and checked by reading, not compiled, not run on hardware
//
#include <hip/hip_runtime.h>
#include <math.h>

// ---------------------------------------------------------------------------
// MI455X (gfx1250) implementation of the MSCA attention block.
//  - 1x1 convs lowered to bf16 WMMA GEMMs (v_wmma_f32_16x16x32_bf16),
//    f32 accumulate. K=32 per issue -> 8x the FLOP/instr of f32 WMMA,
//    keeping the 19.7 GFLOP of GEMM near the ~18us HBM roofline floor.
//  - Depthwise 5x5 + (1x7,7x1)+(1x11,11x1)+(1x21,21x1) fused in one kernel:
//    a 56x56 plane is 12.5KB, three planes (37.6KB) live in LDS (320KB/WGP).
// ---------------------------------------------------------------------------

#define CCH   256
#define HWSZ  3136
#define WIDTH 56
#define BATCH 16
#define NTOT  ((size_t)BATCH * CCH * HWSZ)

typedef __attribute__((ext_vector_type(16))) __bf16 v16bf;
typedef __attribute__((ext_vector_type(8)))  __bf16 v8bf;
typedef __attribute__((ext_vector_type(8)))  float  v8f;
typedef __attribute__((ext_vector_type(8)))  unsigned short u16x8;

__device__ __forceinline__ unsigned short f2bf(float f) {
  unsigned int u = __builtin_bit_cast(unsigned int, f);
  unsigned int r = ((u >> 16) & 1u) + 0x7FFFu;   // round to nearest even
  return (unsigned short)((u + r) >> 16);
}

__device__ __forceinline__ v16bf mk16(v8bf lo, v8bf hi) {
  v16bf r;
#pragma unroll
  for (int i = 0; i < 8; ++i) { r[i] = lo[i]; r[i + 8] = hi[i]; }
  return r;
}

// ---------------------------------------------------------------------------
// Weight fp32 -> bf16 (one-time, 64K elements, stays hot in L2 for all GEMMs)
// ---------------------------------------------------------------------------
__global__ void cvtw_kernel(const float* __restrict__ w,
                            unsigned short* __restrict__ wq) {
  int i = blockIdx.x * 256 + threadIdx.x;
  if (i < CCH * CCH) wq[i] = f2bf(w[i]);
}

// ---------------------------------------------------------------------------
// GEMM: out[o,p] = sum_c W[o,c] * X[c,p] + bias[o]   (per batch image)
// Block: all 256 output rows x 64 pixels. 8 waves, each wave owns
// 2 M-tiles x 4 N-tiles of 16x16 f32 accumulators.
// A fragment (16-bit A 16x32): lane M = lane&15; VGPR0-3 hold K = k0+half*8+0..7,
//   VGPR4-7 hold K = k0+16+half*8+0..7  -> two contiguous 16B global loads.
// B fragment (16-bit B 32x16): lane N = lane&15; same K order as A fragment
//   -> LDS staged [n][k] so it is two contiguous 16B ds loads.
// C/D (f32 16x16): VGPR r -> M = r + 8*half, N = lane&15.
// MODE 0: f32 in, exact-GELU epilogue -> f32 h
// MODE 1: f32 in, (acc+bias)*h epilogue -> bf16 tmp
// MODE 2: bf16 in, plain epilogue -> f32 out
// ---------------------------------------------------------------------------
template <int MODE>
__global__ void __launch_bounds__(256)
gemm_kernel(const unsigned short* __restrict__ Wq,   // 256x256 bf16 [o][c]
            const float* __restrict__ bias,
            const float* __restrict__ Xf,            // f32 input (MODE 0/1)
            const unsigned short* __restrict__ Xq,   // bf16 input (MODE 2)
            const float* __restrict__ Hm,            // h (MODE 1 epilogue)
            float* __restrict__ outF,                // MODE 0/2
            unsigned short* __restrict__ outQ)       // MODE 1
{
  const int tid    = threadIdx.x;
  const int wave   = tid >> 5;
  const int lane   = tid & 31;
  const int lanelo = lane & 15;
  const int half   = lane >> 4;
  const int b      = blockIdx.y;
  const int p0     = blockIdx.x * 64;

  __shared__ __align__(16) unsigned short Bs[64][40];  // [pixel][k], padded rows

  const size_t batchOff = (size_t)b * CCH * HWSZ;

  v8f acc[2][4];
#pragma unroll
  for (int m = 0; m < 2; ++m)
#pragma unroll
    for (int n = 0; n < 4; ++n) acc[m][n] = {};

  const int ks = tid >> 3;        // staging: k row 0..31
  const int n8 = (tid & 7) * 8;   // staging: pixel base 0..56

  for (int k0 = 0; k0 < CCH; k0 += 32) {
    __syncthreads();
    // ---- cooperatively stage B tile (32 k x 64 pixels) transposed ----
    if (MODE == 2) {
      const u16x8 v =
          *(const u16x8*)(Xq + batchOff + (size_t)(k0 + ks) * HWSZ + p0 + n8);
#pragma unroll
      for (int i = 0; i < 8; ++i) Bs[n8 + i][ks] = v[i];
    } else {
      const float4* src =
          (const float4*)(Xf + batchOff + (size_t)(k0 + ks) * HWSZ + p0 + n8);
      float4 v0 = src[0], v1 = src[1];
      Bs[n8 + 0][ks] = f2bf(v0.x); Bs[n8 + 1][ks] = f2bf(v0.y);
      Bs[n8 + 2][ks] = f2bf(v0.z); Bs[n8 + 3][ks] = f2bf(v0.w);
      Bs[n8 + 4][ks] = f2bf(v1.x); Bs[n8 + 5][ks] = f2bf(v1.y);
      Bs[n8 + 6][ks] = f2bf(v1.z); Bs[n8 + 7][ks] = f2bf(v1.w);
    }
    __syncthreads();

    // ---- A fragments from global bf16 weights (L2 resident) ----
    v16bf afr[2];
#pragma unroll
    for (int m = 0; m < 2; ++m) {
      const int Mg = (wave * 2 + m) * 16 + lanelo;
      const unsigned short* row = Wq + (size_t)Mg * CCH + k0 + half * 8;
      v8bf lo = *(const v8bf*)(row);
      v8bf hi = *(const v8bf*)(row + 16);
      afr[m] = mk16(lo, hi);
    }
    // ---- B fragments from LDS ----
    v16bf bfr[4];
#pragma unroll
    for (int n = 0; n < 4; ++n) {
      const unsigned short* row = &Bs[n * 16 + lanelo][half * 8];
      v8bf lo = *(const v8bf*)(row);
      v8bf hi = *(const v8bf*)(row + 16);
      bfr[n] = mk16(lo, hi);
    }
#pragma unroll
    for (int m = 0; m < 2; ++m)
#pragma unroll
      for (int n = 0; n < 4; ++n)
        acc[m][n] = __builtin_amdgcn_wmma_f32_16x16x32_bf16(
            false, afr[m], false, bfr[n], (short)0, acc[m][n], false, false);
  }

  // ---- epilogue ----
#pragma unroll
  for (int m = 0; m < 2; ++m) {
#pragma unroll
    for (int n = 0; n < 4; ++n) {
#pragma unroll
      for (int r = 0; r < 8; ++r) {
        const int Mg = (wave * 2 + m) * 16 + half * 8 + r;
        const int N  = p0 + n * 16 + lanelo;
        const size_t idx = batchOff + (size_t)Mg * HWSZ + N;
        float v = acc[m][n][r] + bias[Mg];
        if (MODE == 0) {
          v = 0.5f * v * (1.0f + erff(v * 0.70710678118654752f));  // exact GELU
          outF[idx] = v;
        } else if (MODE == 1) {
          outQ[idx] = f2bf(v * Hm[idx]);
        } else {
          outF[idx] = v;
        }
      }
    }
  }
}

// ---------------------------------------------------------------------------
// Per-(b,c) plane mean/max reduction for channel attention
// ---------------------------------------------------------------------------
__global__ void __launch_bounds__(256)
pool_kernel(const float* __restrict__ h, float* __restrict__ avg,
            float* __restrict__ mx) {
  const int bc = blockIdx.x;
  const int tid = threadIdx.x;
  const float* hp = h + (size_t)bc * HWSZ;
  float s = 0.0f, m = -3.402823466e38f;
  for (int p = tid; p < HWSZ; p += 256) {
    float v = hp[p];
    s += v;
    m = fmaxf(m, v);
  }
  __shared__ float ss[256], sm[256];
  ss[tid] = s; sm[tid] = m;
  __syncthreads();
  for (int off = 128; off > 0; off >>= 1) {
    if (tid < off) {
      ss[tid] += ss[tid + off];
      sm[tid] = fmaxf(sm[tid], sm[tid + off]);
    }
    __syncthreads();
  }
  if (tid == 0) {
    avg[bc] = ss[0] * (1.0f / (float)HWSZ);
    mx[bc]  = sm[0];
  }
}

// ---------------------------------------------------------------------------
// Channel-attention MLP: att = sig(fc2(relu(fc1(avg)))) + sig(fc2(relu(fc1(max))))
// One block per batch image; tiny matvecs done in LDS.
// ---------------------------------------------------------------------------
__global__ void __launch_bounds__(256)
att_kernel(const float* __restrict__ avg, const float* __restrict__ mx,
           const float* __restrict__ fc1_w, const float* __restrict__ fc1_b,
           const float* __restrict__ fc2_w, const float* __restrict__ fc2_b,
           float* __restrict__ att) {
  const int b = blockIdx.x;
  const int t = threadIdx.x;
  __shared__ float sa[256], sm[256], ha[64], hm[64];
  sa[t] = avg[b * 256 + t];
  sm[t] = mx[b * 256 + t];
  __syncthreads();
  if (t < 64) {
    float s = fc1_b[t];
    for (int j = 0; j < 256; ++j) s += fc1_w[t * 256 + j] * sa[j];
    ha[t] = fmaxf(s, 0.0f);
  } else if (t < 128) {
    const int r = t - 64;
    float s = fc1_b[r];
    for (int j = 0; j < 256; ++j) s += fc1_w[r * 256 + j] * sm[j];
    hm[r] = fmaxf(s, 0.0f);
  }
  __syncthreads();
  float s1 = fc2_b[t], s2 = fc2_b[t];
  for (int j = 0; j < 64; ++j) {
    s1 += fc2_w[t * 64 + j] * ha[j];
    s2 += fc2_w[t * 64 + j] * hm[j];
  }
  att[b * 256 + t] =
      1.0f / (1.0f + expf(-s1)) + 1.0f / (1.0f + expf(-s2));
}

// ---------------------------------------------------------------------------
// Fused depthwise stack: one (b,c) plane per block, everything in LDS.
// ---------------------------------------------------------------------------
template <int K>
__device__ void sep_branch(const float* __restrict__ sI, float* __restrict__ sT,
                           const float* __restrict__ whg, float bh,
                           const float* __restrict__ wvg, float bv,
                           float* acc) {
  const int R = K / 2;
  const int tid = threadIdx.x;
  float wh[K];
#pragma unroll
  for (int i = 0; i < K; ++i) wh[i] = whg[i];
  for (int p = tid; p < HWSZ; p += 256) {
    const int py = p / WIDTH, px = p - py * WIDTH;
    float y = bh;
#pragma unroll
    for (int d = -R; d <= R; ++d) {
      const int xx = px + d;
      if (xx >= 0 && xx < WIDTH) y += wh[d + R] * sI[py * WIDTH + xx];
    }
    sT[p] = y;
  }
  __syncthreads();
  float wv[K];
#pragma unroll
  for (int i = 0; i < K; ++i) wv[i] = wvg[i];
  int it = 0;
  for (int p = tid; p < HWSZ; p += 256, ++it) {
    const int py = p / WIDTH, px = p - py * WIDTH;
    float y = bv;
#pragma unroll
    for (int d = -R; d <= R; ++d) {
      const int yy = py + d;
      if (yy >= 0 && yy < WIDTH) y += wv[d + R] * sT[yy * WIDTH + px];
    }
    acc[it] += y;
  }
  __syncthreads();
}

__global__ void __launch_bounds__(256)
dw_kernel(const float* __restrict__ h, const float* __restrict__ att,
          const float* __restrict__ w5, const float* __restrict__ b5,
          const float* __restrict__ w1x7, const float* __restrict__ b1x7,
          const float* __restrict__ w7x1, const float* __restrict__ b7x1,
          const float* __restrict__ w1x11, const float* __restrict__ b1x11,
          const float* __restrict__ w11x1, const float* __restrict__ b11x1,
          const float* __restrict__ w1x21, const float* __restrict__ b1x21,
          const float* __restrict__ w21x1, const float* __restrict__ b21x1,
          float* __restrict__ xs) {
  const int bc = blockIdx.x;
  const int c  = bc & (CCH - 1);
  const int tid = threadIdx.x;
  __shared__ float sA[HWSZ];   // att-scaled h plane
  __shared__ float sI[HWSZ];   // x_init (after 5x5)
  __shared__ float sT[HWSZ];   // horizontal-conv temp
  const float* hp = h + (size_t)bc * HWSZ;
  const float a = att[bc];
  for (int p = tid; p < HWSZ; p += 256) sA[p] = hp[p] * a;
  __syncthreads();

  float w[25];
#pragma unroll
  for (int i = 0; i < 25; ++i) w[i] = w5[c * 25 + i];
  const float bb = b5[c];
  for (int p = tid; p < HWSZ; p += 256) {
    const int py = p / WIDTH, px = p - py * WIDTH;
    float y = bb;
#pragma unroll
    for (int dy = -2; dy <= 2; ++dy) {
      const int yy = py + dy;
      if (yy < 0 || yy >= WIDTH) continue;
#pragma unroll
      for (int dx = -2; dx <= 2; ++dx) {
        const int xx = px + dx;
        if (xx >= 0 && xx < WIDTH)
          y += w[(dy + 2) * 5 + dx + 2] * sA[yy * WIDTH + xx];
      }
    }
    sI[p] = y;
  }
  __syncthreads();

  float acc[13];
  int it = 0;
  for (int p = tid; p < HWSZ; p += 256, ++it) acc[it] = sI[p];  // x_init term

  sep_branch<7>(sI, sT, w1x7 + c * 7, b1x7[c], w7x1 + c * 7, b7x1[c], acc);
  sep_branch<11>(sI, sT, w1x11 + c * 11, b1x11[c], w11x1 + c * 11, b11x1[c], acc);
  sep_branch<21>(sI, sT, w1x21 + c * 21, b1x21[c], w21x1 + c * 21, b21x1[c], acc);

  float* xp = xs + (size_t)bc * HWSZ;
  it = 0;
  for (int p = tid; p < HWSZ; p += 256, ++it) xp[p] = acc[it];
}

// ---------------------------------------------------------------------------
extern "C" void kernel_launch(void* const* d_in, const int* in_sizes, int n_in,
                              void* d_out, int out_size, void* d_ws,
                              size_t ws_size, hipStream_t stream) {
  const float* x      = (const float*)d_in[0];
  const float* conv_w = (const float*)d_in[1];
  const float* conv_b = (const float*)d_in[2];
  const float* fc1_w  = (const float*)d_in[3];
  const float* fc1_b  = (const float*)d_in[4];
  const float* fc2_w  = (const float*)d_in[5];
  const float* fc2_b  = (const float*)d_in[6];
  const float* w5    = (const float*)d_in[7];
  const float* b5    = (const float*)d_in[8];
  const float* w1x7  = (const float*)d_in[9];
  const float* b1x7  = (const float*)d_in[10];
  const float* w7x1  = (const float*)d_in[11];
  const float* b7x1  = (const float*)d_in[12];
  const float* w1x11 = (const float*)d_in[13];
  const float* b1x11 = (const float*)d_in[14];
  const float* w11x1 = (const float*)d_in[15];
  const float* b11x1 = (const float*)d_in[16];
  const float* w1x21 = (const float*)d_in[17];
  const float* b1x21 = (const float*)d_in[18];
  const float* w21x1 = (const float*)d_in[19];
  const float* b21x1 = (const float*)d_in[20];
  float* out = (float*)d_out;

  char* ws = (char*)d_ws;
  float* h = (float*)ws;                      ws += NTOT * sizeof(float);
  float* xs = (float*)ws;                     ws += NTOT * sizeof(float);
  unsigned short* tmpq = (unsigned short*)ws; ws += NTOT * sizeof(unsigned short);
  unsigned short* wq = (unsigned short*)ws;   ws += (size_t)CCH * CCH * sizeof(unsigned short);
  float* avg = (float*)ws;                    ws += BATCH * CCH * sizeof(float);
  float* mxp = (float*)ws;                    ws += BATCH * CCH * sizeof(float);
  float* att = (float*)ws;                    ws += BATCH * CCH * sizeof(float);

  const dim3 gg(HWSZ / 64, BATCH);  // 49 x 16 pixel-tile blocks per GEMM

  cvtw_kernel<<<(CCH * CCH) / 256, 256, 0, stream>>>(conv_w, wq);

  // h = gelu(conv1x1(x))
  gemm_kernel<0><<<gg, 256, 0, stream>>>(wq, conv_b, x, nullptr, nullptr, h,
                                         nullptr);
  // channel attention
  pool_kernel<<<BATCH * CCH, 256, 0, stream>>>(h, avg, mxp);
  att_kernel<<<BATCH, 256, 0, stream>>>(avg, mxp, fc1_w, fc1_b, fc2_w, fc2_b,
                                        att);
  // xs = x_init + x1 + x2 + x3 over (att * h), fully in LDS
  dw_kernel<<<BATCH * CCH, 256, 0, stream>>>(h, att, w5, b5, w1x7, b1x7, w7x1,
                                             b7x1, w1x11, b1x11, w11x1, b11x1,
                                             w1x21, b1x21, w21x1, b21x1, xs);
  // tmp = (conv1x1(xs)) * h   (stored bf16 for the final GEMM's B operand)
  gemm_kernel<1><<<gg, 256, 0, stream>>>(wq, conv_b, xs, nullptr, h, nullptr,
                                         tmpq);
  // out = conv1x1(tmp)
  gemm_kernel<2><<<gg, 256, 0, stream>>>(wq, conv_b, nullptr, tmpq, nullptr,
                                         out, nullptr);
}
